// MotionEncoderGCN_87720412054214
// MI455X (gfx1250) — compile-verified, matrix operand order checked by reading, not run on hardware
//
#include <hip/hip_runtime.h>
#include <cstdint>
#include <cstddef>

// ---------------- problem constants ----------------
#define NA   20000
#define NL   50000
#define EAA  640000
#define EAL  1000000
#define HDIM 64

#define CDIV(a,b) (((a)+(b)-1)/(b))

// ---------------- WMMA types / helpers ----------------
typedef __attribute__((ext_vector_type(16))) __bf16 bf16x16;
typedef __attribute__((ext_vector_type(8)))  float  v8f;

__device__ inline __bf16 f2bf(float f) {
  unsigned u = __float_as_uint(f);
  u = (u + 0x7FFFu + ((u >> 16) & 1u)) >> 16;   // round-to-nearest-even
  unsigned short s = (unsigned short)u;
  __bf16 r;
  __builtin_memcpy(&r, &s, sizeof(r));
  return r;
}

__device__ inline v8f wmma_bf16(bf16x16 a, bf16x16 b, v8f c) {
  // D = A(16x32 bf16) * B(32x16 bf16) + C(16x16 f32)
  return __builtin_amdgcn_wmma_f32_16x16x32_bf16(
      /*neg_a=*/false, a, /*neg_b=*/false, b,
      /*c_mod=*/(short)0, c, /*reuse_a=*/false, /*reuse_b=*/false);
}

__device__ inline float sigmoidf(float x) { return 1.f / (1.f + __expf(-x)); }

// order-preserving float<->int map for atomicMax on floats
__device__ inline int   f2ord(float f) { int i = __float_as_int(f); return i >= 0 ? i : (i ^ 0x7fffffff); }
__device__ inline float ord2f(int k)   { int i = (k >= 0) ? k : (k ^ 0x7fffffff); return __int_as_float(i); }

// =====================================================================
// LSTM: one wave per 16-row tile. REQUIRES N % 16 == 0 (true for 20000
// and 50000) -> no row guards, no exec-mask churn in the hot loop.
// Whh^T staged once into LDS as bf16 WMMA B-fragments; h kept in LDS in
// *A-fragment layout* as bf16 so each step's A operands are two 32-byte
// LDS vector loads. Gates via v_wmma_f32_16x16x32_bf16 (K=64 -> 2 WMMA).
// mean_mode=0 -> write last h; mean_mode=1 -> write mean over time.
// =====================================================================
template<int T, int D>
__global__ __launch_bounds__(32) void lstm_kernel(
    const float* __restrict__ X,       // [N, T, D]
    const float* __restrict__ Wih,     // [256, D]
    const float* __restrict__ Whh,     // [256, 64]
    const float* __restrict__ bih,     // [256]
    const float* __restrict__ bhh,     // [256]
    float* __restrict__ out,           // [N, 64]
    int mean_mode)
{
  __shared__ bf16x16 sB[32][32];      // [kt*16+nt][lane] : Whh^T B-fragments (32KB)
  __shared__ bf16x16 sHa[32][2];      // h in A-fragment layout (bf16, 2KB)
  __shared__ float   sG[16][256];     // gates
  __shared__ float   sH[16][64];      // hidden (f32, for final output only)
  __shared__ float   sC[16][64];      // cell
  __shared__ float   sAcc[16][64];    // running mean accumulator
  __shared__ float   sX[16][D];       // current-timestep inputs

  const int lane = threadIdx.x & 31;
  const int row0 = blockIdx.x * 16;
  const int nlo  = lane & 15;
  const int khi  = (lane >> 4) << 3;   // upper half-wave holds k%16 in [8,16)
  const int mb   = (lane >> 4) << 3;   // C/D: upper half-wave holds rows 8..15

  // ---- stage B = Whh^T : B[k][n] = Whh[n*64 + k], bf16 fragment layout ----
  for (int kt = 0; kt < 2; ++kt) {
    for (int nt = 0; nt < 16; ++nt) {
      bf16x16 b;
      #pragma unroll
      for (int i = 0; i < 16; ++i) {
        int k = kt * 32 + ((i >> 3) << 4) + khi + (i & 7);
        int n = nt * 16 + nlo;
        b[i] = f2bf(Whh[n * 64 + k]);
      }
      sB[kt * 16 + nt][lane] = b;
    }
  }
  {
    bf16x16 z = {};
    sHa[lane][0] = z;
    sHa[lane][1] = z;
  }
  for (int p = lane; p < 16 * 64; p += 32) {
    sH[p >> 6][p & 63] = 0.f;
    sC[p >> 6][p & 63] = 0.f;
    sAcc[p >> 6][p & 63] = 0.f;
  }
  __syncthreads();

  __bf16* hb = (__bf16*)&sHa[0][0];   // flat view for scattered fragment writes

  for (int t = 0; t < T; ++t) {
    // ---- stage x_t for the 16 rows (once; consumed by all 16 col-tiles) ----
    for (int p = lane; p < 16 * D; p += 32) {
      int m = p / D, d = p - m * D;
      sX[m][d] = X[((size_t)(row0 + m) * T + t) * D + d];
    }
    __syncthreads();

    // ---- A fragments: two 32B LDS vector loads ----
    bf16x16 a0 = sHa[lane][0];
    bf16x16 a1 = sHa[lane][1];

    // ---- gates: 16 tiles of 16 columns, K=64 via 2 WMMAs each ----
    for (int nt = 0; nt < 16; ++nt) {
      const int n = nt * 16 + nlo;
      const float bias = bih[n] + bhh[n];
      float wv[D];
      #pragma unroll
      for (int d = 0; d < D; ++d) wv[d] = Wih[n * D + d];
      v8f acc;
      #pragma unroll
      for (int r = 0; r < 8; ++r) {
        float s = bias;
        #pragma unroll
        for (int d = 0; d < D; ++d) s += sX[mb + r][d] * wv[d];
        acc[r] = s;
      }
      acc = wmma_bf16(a0, sB[nt][lane], acc);
      acc = wmma_bf16(a1, sB[16 + nt][lane], acc);
      #pragma unroll
      for (int r = 0; r < 8; ++r) sG[mb + r][n] = acc[r];
    }
    __syncthreads();
    // ---- elementwise LSTM cell (PyTorch gate order i,f,g,o); writes h
    //      both as f32 (output) and as bf16 A-fragment element ----
    for (int p = lane; p < 16 * 64; p += 32) {
      int m = p >> 6, j = p & 63;
      float gi = sG[m][j], gf = sG[m][64 + j], gg = sG[m][128 + j], go = sG[m][192 + j];
      float c = sigmoidf(gf) * sC[m][j] + sigmoidf(gi) * tanhf(gg);
      float h = sigmoidf(go) * tanhf(c);
      sC[m][j] = c;
      sH[m][j] = h;
      sAcc[m][j] += h;
      // A-layout: lane = m + 16*((j>>3)&1); slot = frag*16 + ((j>>4)&1)*8 + (j&7)
      int tl   = m + (((j >> 3) & 1) << 4);
      int slot = ((j >> 5) << 4) + (((j >> 4) & 1) << 3) + (j & 7);
      hb[tl * 32 + slot] = f2bf(h);
    }
    __syncthreads();
  }

  const float invT = 1.f / (float)T;
  for (int p = lane; p < 16 * 64; p += 32) {
    int m = p >> 6, j = p & 63;
    out[(size_t)(row0 + m) * 64 + j] = mean_mode ? sAcc[m][j] * invT : sH[m][j];
  }
}

// =====================================================================
// Dense GEMM: Y[N,Ko] = X[N,64] @ W[64,Ko], bf16 WMMA, f32 accumulate.
// Ko is a template parameter (64 or 128) so all W/Y strides are
// compile-time immediates (no 64-bit address chains). W is staged once
// per block into LDS in B-fragment layout (<=16KB), shared by 8 waves;
// each tile's B operand is then a single 32-byte LDS vector load.
// REQUIRES N % 16 == 0 (all call sites use 20000 / 50000).
// =====================================================================
template<int Ko>
__global__ __launch_bounds__(256) void gemm64_kernel(
    const float* __restrict__ X, const float* __restrict__ W,
    float* __restrict__ Y, int N)
{
  constexpr int NT = Ko / 16;          // n-tiles
  __shared__ bf16x16 sW[2 * NT][32];   // [kt*NT+nt][lane]

  const int wid  = threadIdx.x >> 5;
  const int lane = threadIdx.x & 31;
  const int nlo  = lane & 15;
  const int khi  = (lane >> 4) << 3;

  // ---- cooperative stage of all B fragments (once per block) ----
  for (int f = wid; f < 2 * NT; f += 8) {
    int kt = f / NT, nt = f - kt * NT;
    bf16x16 b;
    #pragma unroll
    for (int i = 0; i < 16; ++i) {
      int k = kt * 32 + ((i >> 3) << 4) + khi + (i & 7);
      b[i] = f2bf(W[k * Ko + nt * 16 + nlo]);    // immediate-offset loads
    }
    sW[f][lane] = b;
  }
  __syncthreads();

  const int wave  = blockIdx.x * 8 + wid;
  const int tiles = (N >> 4) * NT;
  if (wave >= tiles) return;                     // wave-uniform: EXEC all-ones for WMMA
  const int tm = wave / NT, tn = wave - tm * NT;
  const int m  = tm * 16 + nlo;

  v8f acc = {};
  #pragma unroll
  for (int kt = 0; kt < 2; ++kt) {
    // A fragment: two contiguous 8-float runs -> 4x float4 (32B aligned)
    const float* xp = X + m * 64 + kt * 32 + khi;
    float4 q0 = *(const float4*)(xp);
    float4 q1 = *(const float4*)(xp + 4);
    float4 q2 = *(const float4*)(xp + 16);
    float4 q3 = *(const float4*)(xp + 20);
    bf16x16 a;
    a[0] = f2bf(q0.x); a[1] = f2bf(q0.y); a[2]  = f2bf(q0.z); a[3]  = f2bf(q0.w);
    a[4] = f2bf(q1.x); a[5] = f2bf(q1.y); a[6]  = f2bf(q1.z); a[7]  = f2bf(q1.w);
    a[8] = f2bf(q2.x); a[9] = f2bf(q2.y); a[10] = f2bf(q2.z); a[11] = f2bf(q2.w);
    a[12]= f2bf(q3.x); a[13]= f2bf(q3.y); a[14] = f2bf(q3.z); a[15] = f2bf(q3.w);
    acc = wmma_bf16(a, sW[kt * NT + tn][lane], acc);
  }
  const int mb = (lane >> 4) << 3;
  float* yp = Y + (tm * 16 + mb) * Ko + tn * 16 + nlo;
  #pragma unroll
  for (int r = 0; r < 8; ++r) {
    yp[r * Ko] = acc[r];                         // immediate-offset stores
  }
}

// ---------------- small utility kernels ----------------
__global__ void fill_f_kernel(float* p, float v, int n) {
  int i = blockIdx.x * blockDim.x + threadIdx.x;
  if (i < n) p[i] = v;
}
__global__ void fill_i_kernel(int* p, int v, int n) {
  int i = blockIdx.x * blockDim.x + threadIdx.x;
  if (i < n) p[i] = v;
}
__global__ void rsqrt_kernel(float* p, int n) {
  int i = blockIdx.x * blockDim.x + threadIdx.x;
  if (i < n) p[i] = rsqrtf(p[i]);
}

// edge weights + degree accumulation (dst-segment sum)
__global__ void edge_w_deg_kernel(const float* __restrict__ len, const int* __restrict__ dst,
                                  float* __restrict__ ew, float* __restrict__ deg,
                                  const float* __restrict__ w_e, const float* __restrict__ b_e, int E)
{
  int e = blockIdx.x * blockDim.x + threadIdx.x;
  if (e >= E) return;
  float w = sigmoidf(len[e] * w_e[0] + b_e[0]);
  ew[e] = w;
  atomicAdd(&deg[dst[e]], w);
}

// out[m][j] = b[j] + dinv[m]^2 * xw[m][j]  (GCN self-loop + bias init)
__global__ void gcn_init_kernel(const float* __restrict__ xw, const float* __restrict__ dinv,
                                const float* __restrict__ b, float* __restrict__ out, int N)
{
  int i = blockIdx.x * blockDim.x + threadIdx.x;
  if (i >= N * HDIM) return;
  int m = i >> 6, j = i & 63;
  float di = dinv[m];
  out[i] = b[j] + di * di * xw[i];
}

// grid-stride, one wave per edge: out[dst] += dinv[src]*w*dinv[dst]*xw[src]
// prefetch (global_prefetch_b8) the next edge's source row while this one is in flight
__global__ __launch_bounds__(256) void gcn_scatter_kernel(
    const float* __restrict__ xw, const int* __restrict__ src, const int* __restrict__ dst,
    const float* __restrict__ ew, const float* __restrict__ dinv, float* __restrict__ out, int E)
{
  const int nw   = gridDim.x * (blockDim.x >> 5);
  const int lane = threadIdx.x & 31;
  for (int e = blockIdx.x * (blockDim.x >> 5) + (threadIdx.x >> 5); e < E; e += nw) {
    int en = e + nw;
    if (en < E) __builtin_prefetch(&xw[(size_t)src[en] * 64 + lane * 2], 0, 0);
    int s = src[e], d = dst[e];
    float nrm = dinv[s] * ew[e] * dinv[d];
    atomicAdd(&out[(size_t)d * 64 + lane],      nrm * xw[(size_t)s * 64 + lane]);
    atomicAdd(&out[(size_t)d * 64 + 32 + lane], nrm * xw[(size_t)s * 64 + 32 + lane]);
  }
}

// per-channel mean/var over N rows (64 channels, one block each)
__global__ __launch_bounds__(256) void bn_stats_kernel(const float* __restrict__ x,
                                                       float* __restrict__ stats, int N)
{
  __shared__ float r1[256], r2[256];
  int ch = blockIdx.x;
  float s = 0.f, s2 = 0.f;
  for (int r = threadIdx.x; r < N; r += blockDim.x) {
    float v = x[(size_t)r * 64 + ch];
    s += v; s2 += v * v;
  }
  r1[threadIdx.x] = s; r2[threadIdx.x] = s2;
  __syncthreads();
  for (int o = 128; o > 0; o >>= 1) {
    if (threadIdx.x < o) { r1[threadIdx.x] += r1[threadIdx.x + o]; r2[threadIdx.x] += r2[threadIdx.x + o]; }
    __syncthreads();
  }
  if (threadIdx.x == 0) {
    float m = r1[0] / (float)N;
    stats[ch] = m;
    stats[64 + ch] = r2[0] / (float)N - m * m;
  }
}

__global__ void bn_apply_kernel(float* __restrict__ x, const float* __restrict__ stats,
                                const float* __restrict__ g, const float* __restrict__ b,
                                int N, int do_relu)
{
  int i = blockIdx.x * blockDim.x + threadIdx.x;
  if (i >= N * HDIM) return;
  int ch = i & 63;
  float y = (x[i] - stats[ch]) * rsqrtf(stats[64 + ch] + 1e-5f) * g[ch] + b[ch];
  x[i] = (do_relu && y < 0.f) ? 0.f : y;
}

// a[n*2+h] = sum_c hx[n*Ko + h*ch + c] * att[h*ch + c]
__global__ void att_score_kernel(const float* __restrict__ hx, const float* __restrict__ att,
                                 float* __restrict__ a, int N, int Ko, int ch)
{
  int idx = blockIdx.x * blockDim.x + threadIdx.x;
  if (idx >= N * 2) return;
  int n = idx >> 1, h = idx & 1;
  const float* hp = hx + (size_t)n * Ko + h * ch;
  const float* ap = att + h * ch;
  float s = 0.f;
  for (int c = 0; c < ch; ++c) s += hp[c] * ap[c];
  a[idx] = s;
}

// alpha = leaky_relu(a_s[src]+a_d[dst], 0.2); segment max via int-ordered atomicMax
__global__ void gat_alpha_kernel(const float* __restrict__ a_s, const float* __restrict__ a_d,
                                 const int* __restrict__ src, const int* __restrict__ dst,
                                 float* __restrict__ al, int* __restrict__ amax, int E)
{
  int e = blockIdx.x * blockDim.x + threadIdx.x;
  if (e >= E) return;
  int s = src[e], d = dst[e];
  #pragma unroll
  for (int h = 0; h < 2; ++h) {
    float v = a_s[s * 2 + h] + a_d[d * 2 + h];
    v = (v > 0.f) ? v : 0.2f * v;
    al[(size_t)e * 2 + h] = v;
    atomicMax(&amax[d * 2 + h], f2ord(v));
  }
}

// e = exp(alpha - max); segment sum
__global__ void gat_exp_kernel(float* __restrict__ al, const int* __restrict__ amax,
                               float* __restrict__ ssum, const int* __restrict__ dst, int E)
{
  int e = blockIdx.x * blockDim.x + threadIdx.x;
  if (e >= E) return;
  int d = dst[e];
  #pragma unroll
  for (int h = 0; h < 2; ++h) {
    float v = __expf(al[(size_t)e * 2 + h] - ord2f(amax[d * 2 + h]));
    al[(size_t)e * 2 + h] = v;
    atomicAdd(&ssum[d * 2 + h], v);
  }
}

// grid-stride, one wave per edge: out[dst,:Ko] += hs[src,:Ko]*e/(s+1e-16)
__global__ __launch_bounds__(256) void gat_scatter_kernel(
    const float* __restrict__ hs, const float* __restrict__ al, const float* __restrict__ ssum,
    const int* __restrict__ src, const int* __restrict__ dst, float* __restrict__ out,
    int E, int Ko, int ch)
{
  const int nw   = gridDim.x * (blockDim.x >> 5);
  const int lane = threadIdx.x & 31;
  for (int e = blockIdx.x * (blockDim.x >> 5) + (threadIdx.x >> 5); e < E; e += nw) {
    int en = e + nw;
    if (en < E) __builtin_prefetch(&hs[(size_t)src[en] * Ko + lane * (Ko >> 5)], 0, 0);
    int s = src[e], d = dst[e];
    for (int j = lane; j < Ko; j += 32) {
      int h = j / ch;
      float coef = al[(size_t)e * 2 + h] / (ssum[d * 2 + h] + 1e-16f);
      atomicAdd(&out[(size_t)d * Ko + j], hs[(size_t)s * Ko + j] * coef);
    }
  }
}

// out = acc + bias[j] + res   (GAT1 concat output + residual)
__global__ void add_res_bias_kernel(const float* __restrict__ acc, const float* __restrict__ bias,
                                    const float* __restrict__ res, float* __restrict__ out, int N)
{
  int i = blockIdx.x * blockDim.x + threadIdx.x;
  if (i >= N * HDIM) return;
  out[i] = acc[i] + bias[i & 63] + res[i];
}

// agent_map = mean_over_heads(acc[N,2,64]) + bias[j] + g
__global__ void gat2_final_kernel(const float* __restrict__ acc, const float* __restrict__ bias,
                                  const float* __restrict__ g, float* __restrict__ out, int N)
{
  int i = blockIdx.x * blockDim.x + threadIdx.x;
  if (i >= N * HDIM) return;
  int m = i >> 6, j = i & 63;
  out[i] = 0.5f * (acc[(size_t)m * 128 + j] + acc[(size_t)m * 128 + 64 + j]) + bias[j] + g[i];
}

// =====================================================================
// host-side orchestration
// =====================================================================
extern "C" void kernel_launch(void* const* d_in, const int* in_sizes, int n_in,
                              void* d_out, int out_size, void* d_ws, size_t ws_size,
                              hipStream_t stream)
{
  (void)in_sizes; (void)n_in; (void)out_size; (void)ws_size;

  const float* agent_hist = (const float*)d_in[0];
  const float* lane_nodes = (const float*)d_in[1];
  const float* edges_len  = (const float*)d_in[2];
  const int*   ei_aa      = (const int*)d_in[3];   // [2, EAA] : row0=src, row1=dst
  const int*   ei_al      = (const int*)d_in[4];   // [2, EAL] : row0=agent(dst), row1=lane(src)
  const float* Wih_a = (const float*)d_in[5];
  const float* Whh_a = (const float*)d_in[6];
  const float* bih_a = (const float*)d_in[7];
  const float* bhh_a = (const float*)d_in[8];
  const float* Wih_l = (const float*)d_in[9];
  const float* Whh_l = (const float*)d_in[10];
  const float* bih_l = (const float*)d_in[11];
  const float* bhh_l = (const float*)d_in[12];
  const float* w_e = (const float*)d_in[13];
  const float* b_e = (const float*)d_in[14];
  const float* gcn_W1 = (const float*)d_in[15];
  const float* gcn_b1 = (const float*)d_in[16];
  const float* gcn_bn_g = (const float*)d_in[17];
  const float* gcn_bn_b = (const float*)d_in[18];
  const float* gcn_W2 = (const float*)d_in[19];
  const float* gcn_b2 = (const float*)d_in[20];
  const float* gat_Wsrc1 = (const float*)d_in[21];
  const float* gat_Wdst1 = (const float*)d_in[22];
  const float* gat_as1 = (const float*)d_in[23];
  const float* gat_ad1 = (const float*)d_in[24];
  const float* gat_bias1 = (const float*)d_in[25];
  const float* gat_bn_g = (const float*)d_in[26];
  const float* gat_bn_b = (const float*)d_in[27];
  const float* gat_Wsrc2 = (const float*)d_in[28];
  const float* gat_Wdst2 = (const float*)d_in[29];
  const float* gat_as2 = (const float*)d_in[30];
  const float* gat_ad2 = (const float*)d_in[31];
  const float* gat_bias2 = (const float*)d_in[32];

  float* out          = (float*)d_out;
  float* agent_emb    = out;
  float* agent_social = out + (size_t)NA * HDIM;
  float* agent_map    = out + (size_t)2 * NA * HDIM;
  float* lane_emb     = out + (size_t)3 * NA * HDIM;

  // workspace carve-out (all offsets 256B aligned)
  char*  wsb = (char*)d_ws;
  size_t off = 0;
  auto walloc = [&](size_t bytes) -> void* {
    void* p = wsb + off;
    off = (off + bytes + 255) & ~(size_t)255;
    return p;
  };
  float* ew   = (float*)walloc((size_t)EAA * 4);
  float* deg  = (float*)walloc((size_t)NA * 4);            // becomes dinv in place
  float* xw   = (float*)walloc((size_t)NA * HDIM * 4);
  float* h1   = (float*)walloc((size_t)NA * HDIM * 4);
  float* hsb  = (float*)walloc((size_t)NL * 128 * 4);      // GAT source features (max layer2)
  float* hdb  = (float*)walloc((size_t)NA * 128 * 4);      // GAT dest features
  float* a_s  = (float*)walloc((size_t)NL * 2 * 4);
  float* a_d  = (float*)walloc((size_t)NA * 2 * 4);
  int*   amax = (int*)  walloc((size_t)NA * 2 * 4);
  float* ssum = (float*)walloc((size_t)NA * 2 * 4);
  float* alb  = (float*)walloc((size_t)EAL * 2 * 4);
  float* gbuf = (float*)walloc((size_t)NA * HDIM * 4);
  float* acc  = (float*)walloc((size_t)NA * 128 * 4);
  float* stats= (float*)walloc(128 * 4);

  const int* src_aa = ei_aa;
  const int* dst_aa = ei_aa + EAA;
  const int* src_al = ei_al + EAL;   // lanes (flipped)
  const int* dst_al = ei_al;         // agents

  const int INT_MIN_V = (int)0x80000000;
  const int SCAT_BLOCKS = 2048;      // grid-stride scatter: 16384 waves in flight

  // ---- phase A: LSTMs (WMMA); NA, NL are multiples of 16 ----
  lstm_kernel<20, 5><<<NA / 16, 32, 0, stream>>>(agent_hist, Wih_a, Whh_a, bih_a, bhh_a,
                                                 agent_emb, 0);
  lstm_kernel<10, 2><<<NL / 16, 32, 0, stream>>>(lane_nodes, Wih_l, Whh_l, bih_l, bhh_l,
                                                 lane_emb, 1);

  // ---- phase B: edge weights, degree, GCN x2 ----
  fill_f_kernel<<<CDIV(NA, 256), 256, 0, stream>>>(deg, 1.0f, NA);  // self-loop +1
  edge_w_deg_kernel<<<CDIV(EAA, 256), 256, 0, stream>>>(edges_len, dst_aa, ew, deg, w_e, b_e, EAA);
  rsqrt_kernel<<<CDIV(NA, 256), 256, 0, stream>>>(deg, NA);          // deg -> dinv

  gemm64_kernel<64><<<CDIV((NA / 16) * 4, 8), 256, 0, stream>>>(agent_emb, gcn_W1, xw, NA);
  gcn_init_kernel<<<CDIV(NA * HDIM, 256), 256, 0, stream>>>(xw, deg, gcn_b1, h1, NA);
  gcn_scatter_kernel<<<SCAT_BLOCKS, 256, 0, stream>>>(xw, src_aa, dst_aa, ew, deg, h1, EAA);
  bn_stats_kernel<<<64, 256, 0, stream>>>(h1, stats, NA);
  bn_apply_kernel<<<CDIV(NA * HDIM, 256), 256, 0, stream>>>(h1, stats, gcn_bn_g, gcn_bn_b, NA, 1);

  gemm64_kernel<64><<<CDIV((NA / 16) * 4, 8), 256, 0, stream>>>(h1, gcn_W2, xw, NA);
  gcn_init_kernel<<<CDIV(NA * HDIM, 256), 256, 0, stream>>>(xw, deg, gcn_b2, agent_social, NA);
  gcn_scatter_kernel<<<SCAT_BLOCKS, 256, 0, stream>>>(xw, src_aa, dst_aa, ew, deg, agent_social, EAA);

  // ---- phase C: GAT layer 1 (heads=2, ch=32, concat) ----
  gemm64_kernel<64><<<CDIV((NL / 16) * 4, 8), 256, 0, stream>>>(lane_emb, gat_Wsrc1, hsb, NL);
  gemm64_kernel<64><<<CDIV((NA / 16) * 4, 8), 256, 0, stream>>>(agent_social, gat_Wdst1, hdb, NA);
  att_score_kernel<<<CDIV(NL * 2, 256), 256, 0, stream>>>(hsb, gat_as1, a_s, NL, 64, 32);
  att_score_kernel<<<CDIV(NA * 2, 256), 256, 0, stream>>>(hdb, gat_ad1, a_d, NA, 64, 32);
  fill_i_kernel<<<CDIV(NA * 2, 256), 256, 0, stream>>>(amax, INT_MIN_V, NA * 2);
  fill_f_kernel<<<CDIV(NA * 2, 256), 256, 0, stream>>>(ssum, 0.f, NA * 2);
  gat_alpha_kernel<<<CDIV(EAL, 256), 256, 0, stream>>>(a_s, a_d, src_al, dst_al, alb, amax, EAL);
  gat_exp_kernel<<<CDIV(EAL, 256), 256, 0, stream>>>(alb, amax, ssum, dst_al, EAL);
  fill_f_kernel<<<CDIV(NA * 64, 256), 256, 0, stream>>>(acc, 0.f, NA * 64);
  gat_scatter_kernel<<<SCAT_BLOCKS, 256, 0, stream>>>(hsb, alb, ssum, src_al, dst_al, acc, EAL, 64, 32);
  add_res_bias_kernel<<<CDIV(NA * HDIM, 256), 256, 0, stream>>>(acc, gat_bias1, agent_social, gbuf, NA);
  bn_stats_kernel<<<64, 256, 0, stream>>>(gbuf, stats, NA);
  bn_apply_kernel<<<CDIV(NA * HDIM, 256), 256, 0, stream>>>(gbuf, stats, gat_bn_g, gat_bn_b, NA, 1);

  // ---- phase D: GAT layer 2 (heads=2, ch=64, mean) ----
  gemm64_kernel<128><<<CDIV((NL / 16) * 8, 8), 256, 0, stream>>>(lane_emb, gat_Wsrc2, hsb, NL);
  gemm64_kernel<128><<<CDIV((NA / 16) * 8, 8), 256, 0, stream>>>(gbuf, gat_Wdst2, hdb, NA);
  att_score_kernel<<<CDIV(NL * 2, 256), 256, 0, stream>>>(hsb, gat_as2, a_s, NL, 128, 64);
  att_score_kernel<<<CDIV(NA * 2, 256), 256, 0, stream>>>(hdb, gat_ad2, a_d, NA, 128, 64);
  fill_i_kernel<<<CDIV(NA * 2, 256), 256, 0, stream>>>(amax, INT_MIN_V, NA * 2);
  fill_f_kernel<<<CDIV(NA * 2, 256), 256, 0, stream>>>(ssum, 0.f, NA * 2);
  gat_alpha_kernel<<<CDIV(EAL, 256), 256, 0, stream>>>(a_s, a_d, src_al, dst_al, alb, amax, EAL);
  gat_exp_kernel<<<CDIV(EAL, 256), 256, 0, stream>>>(alb, amax, ssum, dst_al, EAL);
  fill_f_kernel<<<CDIV(NA * 128, 256), 256, 0, stream>>>(acc, 0.f, NA * 128);
  gat_scatter_kernel<<<SCAT_BLOCKS, 256, 0, stream>>>(hsb, alb, ssum, src_al, dst_al, acc, EAL, 128, 64);
  gat2_final_kernel<<<CDIV(NA * HDIM, 256), 256, 0, stream>>>(acc, gat_bias2, gbuf, agent_map, NA);
}
